// AttentionCell_89232240541705
// MI455X (gfx1250) — compile-verified, matrix operand order checked by reading, not run on hardware
//
#include <hip/hip_runtime.h>
#include <hip/hip_bf16.h>
#include <cstdint>

#define B_    64
#define L_    1024
#define E_    80
#define H_    512
#define DE_   512
#define V_    500
#define FOURH 2048

typedef __bf16 bf16_t;
typedef __attribute__((ext_vector_type(16))) bf16_t v16bf;
typedef __attribute__((ext_vector_type(8)))  float  v8f;
typedef __attribute__((ext_vector_type(2)))  float  v2f;

__device__ __forceinline__ float sigmoidf_(float x) { return 1.0f / (1.0f + __expf(-x)); }

// ---------------------------------------------------------------------------
// Kernel 0: pack W1e (512x512 f32, row-major [k][n]) into bf16 B-fragment
// layout: W1p[kt][nt][lane][v], klocal = (v&7) + ((v>>3)<<4) + ((lane>>4)<<3)
// ---------------------------------------------------------------------------
__global__ void pack_w1e_kernel(const float* __restrict__ W1e, bf16_t* __restrict__ W1p) {
    unsigned i = blockIdx.x * 256u + threadIdx.x;
    if (i >= 512u * 512u) return;
    unsigned v    = i & 15u;
    unsigned lane = (i >> 4) & 31u;
    unsigned nt   = (i >> 9) & 31u;
    unsigned kt   = i >> 14;
    unsigned klocal = (v & 7u) + ((v >> 3) << 4) + ((lane >> 4) << 3);
    unsigned k = kt * 32u + klocal;
    unsigned n = nt * 16u + (lane & 15u);
    W1p[i] = (bf16_t)W1e[k * 512u + n];
}

// ---------------------------------------------------------------------------
// Kernel 1: z = concat(inputs, o_prev) @ Wk + h_prev @ Uk + b  via f32 WMMA
// Grid: 64 blocks x 256 thr = 512 waves; wave -> one 16x16 tile of (64,2048).
// Concat K-space split into 3 exact segments (all multiples of 4) so each
// inner loop is branch-free: float2 A load + 2 B loads + 1 WMMA per step.
// ---------------------------------------------------------------------------
__global__ void lstm_z_kernel(const float* __restrict__ inputs, const float* __restrict__ o_prev,
                              const float* __restrict__ h_prev, const float* __restrict__ Wk,
                              const float* __restrict__ Uk, const float* __restrict__ bias,
                              float* __restrict__ zbuf) {
    int wid  = (blockIdx.x * blockDim.x + threadIdx.x) >> 5;   // 0..511
    int lane = threadIdx.x & 31;
    int mt = wid >> 7;           // 0..3   (M tiles of 16 over B=64)
    int nt = wid & 127;          // 0..127 (N tiles of 16 over 2048)
    int gm = mt * 16 + (lane & 15);
    int n  = nt * 16 + (lane & 15);
    int kh = (lane >> 4) * 2;    // A/B k-offset per half-wave (32-bit 16x4 layout)

    v8f c = {};

    // --- segment 1: k in [0,80): A = inputs, B = Wk rows [0,80) ---
    {
        const float* ar = inputs + (size_t)gm * E_ + kh;
        const float* br = Wk + (size_t)kh * FOURH + n;
        for (int kb = 0; kb < E_; kb += 4) {
            float2 f2 = *(const float2*)(ar + kb);
            v2f a; a.x = f2.x; a.y = f2.y;
            v2f bb; bb.x = br[(size_t)kb * FOURH];
                    bb.y = br[(size_t)(kb + 1) * FOURH];
            c = __builtin_amdgcn_wmma_f32_16x16x4_f32(false, a, false, bb, (short)0, c, false, false);
        }
    }
    // --- segment 2: k in [80,592): A = o_prev, B = Wk rows [80,592) ---
    {
        const float* ar = o_prev + (size_t)gm * H_ + kh;
        const float* br = Wk + (size_t)(E_ + kh) * FOURH + n;
        for (int kb = 0; kb < H_; kb += 4) {
            float2 f2 = *(const float2*)(ar + kb);
            v2f a; a.x = f2.x; a.y = f2.y;
            v2f bb; bb.x = br[(size_t)kb * FOURH];
                    bb.y = br[(size_t)(kb + 1) * FOURH];
            c = __builtin_amdgcn_wmma_f32_16x16x4_f32(false, a, false, bb, (short)0, c, false, false);
        }
    }
    // --- segment 3: k in [592,1104): A = h_prev, B = Uk ---
    {
        const float* ar = h_prev + (size_t)gm * H_ + kh;
        const float* br = Uk + (size_t)kh * FOURH + n;
        for (int kb = 0; kb < H_; kb += 4) {
            float2 f2 = *(const float2*)(ar + kb);
            v2f a; a.x = f2.x; a.y = f2.y;
            v2f bb; bb.x = br[(size_t)kb * FOURH];
                    bb.y = br[(size_t)(kb + 1) * FOURH];
            c = __builtin_amdgcn_wmma_f32_16x16x4_f32(false, a, false, bb, (short)0, c, false, false);
        }
    }

    int lhi = lane >> 4;
#pragma unroll
    for (int r = 0; r < 8; ++r) {
        int row = mt * 16 + r + lhi * 8;
        int col = nt * 16 + (lane & 15);
        zbuf[(size_t)row * FOURH + col] = c[r] + bias[col];
    }
}

// ---------------------------------------------------------------------------
// Kernel 2: gates -> c_new, h
// ---------------------------------------------------------------------------
__global__ void gates_kernel(const float* __restrict__ zbuf, const float* __restrict__ c_prev,
                             float* __restrict__ h_out, float* __restrict__ c_out) {
    int gid = blockIdx.x * 256 + threadIdx.x;
    if (gid >= B_ * H_) return;
    int b = gid >> 9, j = gid & 511;
    const float* zr = zbuf + (size_t)b * FOURH;
    float ig = sigmoidf_(zr[j]);
    float fg = sigmoidf_(zr[H_ + j]);
    float gg = tanhf(zr[2 * H_ + j]);
    float og = sigmoidf_(zr[3 * H_ + j]);
    float cn = fg * c_prev[gid] + ig * gg;
    c_out[gid] = cn;
    h_out[gid] = og * tanhf(cn);
}

// ---------------------------------------------------------------------------
// Kernel 3: hW2h[b,d] = h[b,:] @ W2h[:,d]
// ---------------------------------------------------------------------------
__global__ void hw2h_kernel(const float* __restrict__ h, const float* __restrict__ W2h,
                            float* __restrict__ hW2h) {
    int gid = blockIdx.x * 256 + threadIdx.x;
    if (gid >= B_ * 512) return;
    int b = gid >> 9, d = gid & 511;
    float acc = 0.f;
    for (int k = 0; k < H_; ++k) acc += h[b * H_ + k] * W2h[(size_t)k * 512 + d];
    hW2h[gid] = acc;
}

// ---------------------------------------------------------------------------
// Kernel 4: fused attention scores (THE big GEMM, bf16 WMMA, f32 accum)
// scores[b,l] = sum_d tanh((seq[b]@W1e)[l,d] + hW2h[b,d]) * beta[d]
// Grid: (L/32, B), 256 threads = 8 waves (2 M-subtiles x 4 N-quadrants)
// ---------------------------------------------------------------------------
__global__ void attn_scores_kernel(const float* __restrict__ seq, const bf16_t* __restrict__ W1p,
                                   const float* __restrict__ hW2h, const float* __restrict__ beta,
                                   float* __restrict__ scores) {
    const int b     = blockIdx.y;
    const int lbase = blockIdx.x * 32;
    const int wave  = threadIdx.x >> 5;
    const int lane  = threadIdx.x & 31;
    const int mi    = wave >> 2;     // 0..1
    const int nq    = wave & 3;      // 0..3
    const int laneHi = lane >> 4;
    const int mrow  = lbase + mi * 16 + (lane & 15);

    __shared__ float s_lds[32];
    if (threadIdx.x < 32) s_lds[threadIdx.x] = 0.0f;
    __syncthreads();

    v8f c[8];
    {
        v8f z = {};
#pragma unroll
        for (int t = 0; t < 8; ++t) c[t] = z;
    }

    const float* arow = seq + ((size_t)(b * L_ + mrow)) * DE_;

    for (int kt = 0; kt < 16; ++kt) {
        if (kt < 15)
            __builtin_prefetch((const void*)(arow + (kt + 1) * 32), 0, 3);
        // ---- A fragment: 16 bf16 per lane (16-bit 16x32 layout) ----
        v16bf a;
        const float* ap = arow + kt * 32 + laneHi * 8;
#pragma unroll
        for (int vp = 0; vp < 8; ++vp) {
            int klocal = (vp & 3) * 2 + (vp >> 2) * 16;   // pairs 0,2,4,6,16,18,20,22
            float2 f2 = *(const float2*)(ap + klocal);
            a[vp * 2]     = (bf16_t)f2.x;
            a[vp * 2 + 1] = (bf16_t)f2.y;
        }
        // ---- 8 B tiles: contiguous pre-packed bf16 fragments ----
        const bf16_t* bp = W1p + (((size_t)kt * 32 + nq * 8) * 32 + lane) * 16;
#pragma unroll
        for (int t = 0; t < 8; ++t) {
            v16bf bfrag = *(const v16bf*)(bp + (size_t)t * 32 * 16);
            c[t] = __builtin_amdgcn_wmma_f32_16x16x32_bf16(false, a, false, bfrag,
                                                           (short)0, c[t], false, false);
        }
    }

    // ---- fused tanh(.) * beta reduction over this wave's 128 columns ----
    float s[8];
#pragma unroll
    for (int r = 0; r < 8; ++r) s[r] = 0.f;
    const float* hw = hW2h + b * 512;
#pragma unroll
    for (int t = 0; t < 8; ++t) {
        int n = nq * 128 + t * 16 + (lane & 15);
        float add = hw[n];
        float bw  = beta[n];
#pragma unroll
        for (int r = 0; r < 8; ++r)
            s[r] += tanhf(c[t][r] + add) * bw;
    }
    // reduce over the 16 lanes of each half-wave (n dimension)
#pragma unroll
    for (int off = 1; off < 16; off <<= 1) {
#pragma unroll
        for (int r = 0; r < 8; ++r) s[r] += __shfl_xor(s[r], off, 32);
    }
    if ((lane & 15) == 0) {
        int mb = mi * 16 + laneHi * 8;
#pragma unroll
        for (int r = 0; r < 8; ++r) atomicAdd(&s_lds[mb + r], s[r]);
    }
    __syncthreads();
    if (threadIdx.x < 32)
        scores[(size_t)b * L_ + lbase + threadIdx.x] = s_lds[threadIdx.x];
}

// ---------------------------------------------------------------------------
// Kernel 5: softmax over L (in-place), one block per batch row
// ---------------------------------------------------------------------------
__global__ void softmax_kernel(float* __restrict__ scores) {
    int b = blockIdx.x, tid = threadIdx.x;
    float* row = scores + (size_t)b * L_;
    float v[4], m = -1e30f;
#pragma unroll
    for (int i = 0; i < 4; ++i) { v[i] = row[tid + i * 256]; m = fmaxf(m, v[i]); }
    __shared__ float swork[8];
    __shared__ float sbcast;
#pragma unroll
    for (int off = 16; off; off >>= 1) m = fmaxf(m, __shfl_xor(m, off, 32));
    if ((tid & 31) == 0) swork[tid >> 5] = m;
    __syncthreads();
    if (tid == 0) { float mm = swork[0]; for (int i = 1; i < 8; ++i) mm = fmaxf(mm, swork[i]); sbcast = mm; }
    __syncthreads();
    m = sbcast;
    float s = 0.f;
#pragma unroll
    for (int i = 0; i < 4; ++i) { v[i] = __expf(v[i] - m); s += v[i]; }
#pragma unroll
    for (int off = 16; off; off >>= 1) s += __shfl_xor(s, off, 32);
    __syncthreads();            // protect swork reuse
    if ((tid & 31) == 0) swork[tid >> 5] = s;
    __syncthreads();
    if (tid == 0) { float ss = 0.f; for (int i = 0; i < 8; ++i) ss += swork[i]; sbcast = ss; }
    __syncthreads();
    float inv = 1.0f / sbcast;
#pragma unroll
    for (int i = 0; i < 4; ++i) row[tid + i * 256] = v[i] * inv;
}

// ---------------------------------------------------------------------------
// Kernel 6: ctx[b,d] = sum_l alpha[b,l] * seq[b,l,d]   (memory-bound, coalesced)
// ---------------------------------------------------------------------------
__global__ void ctx_kernel(const float* __restrict__ alpha, const float* __restrict__ seq,
                           float* __restrict__ ctx) {
    int b = blockIdx.y;
    int d = blockIdx.x * 128 + threadIdx.x;
    const float* ar = alpha + (size_t)b * L_;
    const float* sr = seq + (size_t)b * L_ * DE_ + d;
    float acc = 0.f;
    for (int l = 0; l < L_; ++l) acc += ar[l] * sr[(size_t)l * DE_];
    ctx[(size_t)b * DE_ + d] = acc;
}

// ---------------------------------------------------------------------------
// Kernel 7: new_o = tanh(concat(h, ctx) @ W3o)
// ---------------------------------------------------------------------------
__global__ void newo_kernel(const float* __restrict__ h, const float* __restrict__ ctx,
                            const float* __restrict__ W3o, float* __restrict__ new_o) {
    int gid = blockIdx.x * 256 + threadIdx.x;
    if (gid >= B_ * 512) return;
    int b = gid >> 9, j = gid & 511;
    float acc = 0.f;
    for (int k = 0; k < H_; ++k)   acc += h[b * H_ + k]   * W3o[(size_t)k * 512 + j];
    for (int k = 0; k < DE_; ++k)  acc += ctx[b * DE_ + k] * W3o[(size_t)(H_ + k) * 512 + j];
    new_o[gid] = tanhf(acc);
}

// ---------------------------------------------------------------------------
// Kernel 8: logits = new_o @ W4
// ---------------------------------------------------------------------------
__global__ void logits_kernel(const float* __restrict__ new_o, const float* __restrict__ W4,
                              float* __restrict__ logits) {
    int gid = blockIdx.x * 256 + threadIdx.x;
    if (gid >= B_ * V_) return;
    int b = gid / V_, vv = gid % V_;
    float acc = 0.f;
    for (int k = 0; k < 512; ++k) acc += new_o[b * 512 + k] * W4[(size_t)k * V_ + vv];
    logits[gid] = acc;
}

// ---------------------------------------------------------------------------
extern "C" void kernel_launch(void* const* d_in, const int* in_sizes, int n_in,
                              void* d_out, int out_size, void* d_ws, size_t ws_size,
                              hipStream_t stream) {
    const float* inputs = (const float*)d_in[0];
    const float* seq    = (const float*)d_in[1];
    const float* h_prev = (const float*)d_in[2];
    const float* c_prev = (const float*)d_in[3];
    const float* o_prev = (const float*)d_in[4];
    const float* Wk     = (const float*)d_in[5];
    const float* Uk     = (const float*)d_in[6];
    const float* bias   = (const float*)d_in[7];
    const float* W1e    = (const float*)d_in[8];
    const float* W2h    = (const float*)d_in[9];
    const float* beta_w = (const float*)d_in[10];
    const float* W3o    = (const float*)d_in[11];
    const float* W4     = (const float*)d_in[12];

    float* out = (float*)d_out;
    float* logits = out;                       // 64*500
    float* h_out  = out + B_ * V_;             // 64*512
    float* c_out  = h_out + B_ * H_;           // 64*512
    float* o_out  = c_out + B_ * H_;           // 64*512

    char* ws = (char*)d_ws;
    float*  zbuf   = (float*)(ws);                          // 64*2048 f32  (512 KB)
    float*  hW2h   = (float*)(ws + 524288);                 // 64*512  f32  (128 KB)
    float*  scores = (float*)(ws + 524288 + 131072);        // 64*1024 f32  (256 KB)
    float*  ctx    = (float*)(ws + 524288 + 131072 + 262144);      // 64*512 f32 (128 KB)
    bf16_t* W1p    = (bf16_t*)(ws + 524288 + 131072 + 262144 + 131072); // 512*512 bf16 (512 KB)

    // 0) pack W1e -> bf16 fragment layout (independent, launch first)
    pack_w1e_kernel<<<(512 * 512 + 255) / 256, 256, 0, stream>>>(W1e, W1p);
    // 1) LSTM pre-activations via f32 WMMA
    lstm_z_kernel<<<64, 256, 0, stream>>>(inputs, o_prev, h_prev, Wk, Uk, bias, zbuf);
    // 2) gates
    gates_kernel<<<(B_ * H_ + 255) / 256, 256, 0, stream>>>(zbuf, c_prev, h_out, c_out);
    // 3) hW2h
    hw2h_kernel<<<(B_ * 512 + 255) / 256, 256, 0, stream>>>(h_out, W2h, hW2h);
    // 4) fused attention scores (bf16 WMMA)
    {
        dim3 grid(L_ / 32, B_);
        attn_scores_kernel<<<grid, 256, 0, stream>>>(seq, W1p, hW2h, beta_w, scores);
    }
    // 5) softmax over L
    softmax_kernel<<<B_, 256, 0, stream>>>(scores);
    // 6) ctx = alpha @ seq
    {
        dim3 grid(DE_ / 128, B_);
        ctx_kernel<<<grid, 128, 0, stream>>>(scores, seq, ctx);
    }
    // 7) new_o
    newo_kernel<<<(B_ * 512 + 255) / 256, 256, 0, stream>>>(h_out, ctx, W3o, o_out);
    // 8) logits
    logits_kernel<<<(B_ * V_ + 255) / 256, 256, 0, stream>>>(o_out, W4, logits);
}
